// csFSMNCell_88562225643918
// MI455X (gfx1250) — matrix operand order repaired and checked by reading, not compiled, not run on hardware
//
#include <hip/hip_runtime.h>

// ---------------------------------------------------------------------------
// csFSMN cell for MI455X (gfx1250, wave32, WMMA, async-to-LDS)
//   P  = bf16( pad_mask * (X @ W1 + b1) )       [GEMM1, v_wmma_f32_16x16x32_bf16]
//   P2 = banded-conv_T(P) (weights 1, 1..30)    [O(1) recurrence per step]
//   Out= f32( P2 @ W2 + b2 )                    [GEMM2, async bf16 A-tiles]
// Shapes: B=8, T=2048, D=1024  ->  M=16384, N=K=1024
// Block tile 256x128, 8 wave32 waves, wave tile 64x64 (4x4 WMMA accs),
// double-buffered LDS + register/async prefetch.
// ---------------------------------------------------------------------------

typedef __attribute__((ext_vector_type(16))) __bf16 v16bf;
typedef __attribute__((ext_vector_type(8)))  float  v8f;

__device__ __forceinline__ unsigned short f2bf(float f) {
  unsigned u = __float_as_uint(f);
  u += 0x7FFFu + ((u >> 16) & 1u);   // round-to-nearest-even
  return (unsigned short)(u >> 16);
}
__device__ __forceinline__ float bf2f(unsigned short h) {
  return __uint_as_float(((unsigned)h) << 16);
}
__device__ __forceinline__ unsigned pack2bf(float lo, float hi) {
  return (unsigned)f2bf(lo) | ((unsigned)f2bf(hi) << 16);
}

union FragBF {
  uint4 u[2];
  v16bf v;
};

#define LDP 40   // LDS row stride in ushorts (80 B: 16-B aligned, bank-skewed)

// MODE 0: A = f32 X (reg prefetch + convert), epilogue = *pad_mask -> bf16 P
// MODE 1: A = bf16 P2 (global_load_async_to_lds_b128), epilogue = f32 Out
template <int MODE>
__global__ __launch_bounds__(256) void fsmn_gemm(
    const float* __restrict__ Af32, const unsigned short* __restrict__ Abf,
    const float* __restrict__ W, const float* __restrict__ bias,
    const float* __restrict__ pad, unsigned short* __restrict__ outBf,
    float* __restrict__ outF) {
  constexpr int KD = 1024, ND = 1024;
  constexpr int NK = KD / 32;

  __shared__ __align__(16) unsigned short As[2][256 * LDP];  // [m][k] bf16
  __shared__ __align__(16) unsigned short Bs[2][128 * LDP];  // [n][k] bf16 (transposed)

  const int tid  = threadIdx.x;
  const int lane = tid & 31;
  const int wave = tid >> 5;
  const int wrow = wave >> 1;  // 0..3 : 64 M-rows per wave
  const int wcol = wave & 1;   // 0..1 : 64 N-cols per wave
  const int lm   = lane & 15;
  const int hi   = lane >> 4;

  const int m0 = blockIdx.y * 256;
  const int n0 = blockIdx.x * 128;

  v8f acc[4][4];
#pragma unroll
  for (int f = 0; f < 4; ++f)
#pragma unroll
    for (int g = 0; g < 4; ++g) acc[f][g] = (v8f)(0.0f);

  float4 pa[8];      // MODE0 A prefetch regs (256x32 f32 tile / 256 thr)
  float4 pb[2][2];   // W prefetch regs: 2 items x (2 adjacent k-rows)

  // ---- global -> regs (A mode0 + B), and async A (mode1) for K-chunk k0 ----
  auto load_regs = [&](int k0) {
    if (MODE == 0) {
#pragma unroll
      for (int i = 0; i < 8; ++i) {
        int lin = tid + i * 256;  // 0..2047
        int row = lin >> 3;
        int c4  = (lin & 7) * 4;
        pa[i] = *(const float4*)(Af32 + (size_t)(m0 + row) * KD + k0 + c4);
      }
    }
#pragma unroll
    for (int i = 0; i < 2; ++i) {
      int lin = tid + i * 256;     // 0..511
      int krr = lin >> 5;          // 0..15 (pair of k-rows)
      int n4  = (lin & 31) * 4;    // 0..124
      pb[i][0] = *(const float4*)(W + (size_t)(k0 + 2 * krr + 0) * ND + n0 + n4);
      pb[i][1] = *(const float4*)(W + (size_t)(k0 + 2 * krr + 1) * ND + n0 + n4);
    }
  };

  auto issue_async = [&](int k0, int bx) {
#pragma unroll
    for (int i = 0; i < 4; ++i) {
      int lin = tid + i * 256;  // 0..1023 (256 rows x 4 chunks of 16B)
      int row = lin >> 2;
      int c8  = (lin & 3) * 8;
      const unsigned short* g = Abf + (size_t)(m0 + row) * KD + k0 + c8;
      unsigned lds = (unsigned)(unsigned long long)(&As[bx][row * LDP + c8]);
      asm volatile("global_load_async_to_lds_b128 %0, %1, off"
                   :: "v"(lds), "v"(g) : "memory");
    }
  };

  // ---- regs -> LDS buffer bx ----
  auto store_lds = [&](int bx) {
    if (MODE == 0) {
#pragma unroll
      for (int i = 0; i < 8; ++i) {
        int lin = tid + i * 256;
        int row = lin >> 3;
        int c4  = (lin & 7) * 4;
        ushort4 h;
        h.x = f2bf(pa[i].x); h.y = f2bf(pa[i].y);
        h.z = f2bf(pa[i].z); h.w = f2bf(pa[i].w);
        *(ushort4*)(&As[bx][row * LDP + c4]) = h;
      }
    }
#pragma unroll
    for (int i = 0; i < 2; ++i) {
      int lin = tid + i * 256;
      int krr = lin >> 5;
      int n4  = (lin & 31) * 4;
      const float* lo = (const float*)&pb[i][0];
      const float* hic = (const float*)&pb[i][1];
#pragma unroll
      for (int j = 0; j < 4; ++j)
        *(unsigned*)(&Bs[bx][(n4 + j) * LDP + krr * 2]) = pack2bf(lo[j], hic[j]);
    }
  };

  // ---- pipeline ----
  load_regs(0);
  if constexpr (MODE == 1) issue_async(0, 0);
  store_lds(0);

  for (int ks = 0; ks < NK; ++ks) {
    const int cur = ks & 1;
    if constexpr (MODE == 1)
      asm volatile("s_wait_asynccnt 0x0" ::: "memory");
    __syncthreads();

    const bool more = (ks + 1) < NK;
    if (more) {
      load_regs((ks + 1) * 32);
      if constexpr (MODE == 1) issue_async((ks + 1) * 32, cur ^ 1);
    }

    // A 16x32 bf16 lane layout: lanes 0-15 hold K in [0,8)+[16,24); lanes 16-31: [8,16)+[24,32)
    FragBF a[4];
#pragma unroll
    for (int f = 0; f < 4; ++f) {
      int row = wrow * 64 + f * 16 + lm;
      const unsigned short* p = &As[cur][row * LDP + hi * 8];
      a[f].u[0] = *(const uint4*)(p);
      a[f].u[1] = *(const uint4*)(p + 16);
    }
    // B 32x16 bf16 lane layout: lanes 0-15 hold K in [0,8)+[16,24); lanes 16-31: [8,16)+[24,32)
    FragBF b[4];
#pragma unroll
    for (int g = 0; g < 4; ++g) {
      int nrow = wcol * 64 + g * 16 + lm;
      const unsigned short* p = &Bs[cur][nrow * LDP + hi * 8];
      b[g].u[0] = *(const uint4*)(p);
      b[g].u[1] = *(const uint4*)(p + 16);
    }
#pragma unroll
    for (int f = 0; f < 4; ++f)
#pragma unroll
      for (int g = 0; g < 4; ++g)
        acc[f][g] = __builtin_amdgcn_wmma_f32_16x16x32_bf16(
            false, a[f].v, false, b[g].v, (short)0, acc[f][g], false, false);

    if (more) store_lds(cur ^ 1);
  }

  // ---- epilogue ----
  // C/D layout: VGPR v -> M = v (lanes 0-15) or v+8 (lanes 16-31); N = lane%16
#pragma unroll
  for (int g = 0; g < 4; ++g) {
    int col = n0 + wcol * 64 + g * 16 + lm;
    float bc = bias[col];
#pragma unroll
    for (int f = 0; f < 4; ++f) {
      int rbase = m0 + wrow * 64 + f * 16 + hi * 8;
#pragma unroll
      for (int v = 0; v < 8; ++v) {
        int row = rbase + v;
        float val = acc[f][g][v] + bc;
        if (MODE == 0) {
          float mv = pad[row];  // pad_mask flat [B*T] == row index
          outBf[(size_t)row * ND + col] = f2bf(val * mv);
        } else {
          outF[(size_t)row * ND + col] = val;
        }
      }
    }
  }
}

// ---------------------------------------------------------------------------
// Banded memory matmul as O(1)-per-step recurrence along T:
//   out[t] = x[t] + A[t],  A[t] = sum_{d=1..30} d*x[t-d]
//   B[t] = B[t-1] + x[t-1] - x[t-31] ; A[t] = A[t-1] + B[t] - 30*x[t-31]
// ---------------------------------------------------------------------------
__global__ __launch_bounds__(256) void banded_conv(
    const unsigned short* __restrict__ P, unsigned short* __restrict__ P2) {
  constexpr int T = 2048, D = 1024, CH = 128;
  const int c  = blockIdx.x * 256 + threadIdx.x;
  const int t0 = blockIdx.y * CH;
  const int b  = blockIdx.z;
  const size_t base = (size_t)b * T * D + c;

  float Bsum = 0.0f, Asum = 0.0f;
  const int dmax = (t0 < 30) ? t0 : 30;
  for (int d = 1; d <= dmax; ++d) {
    float x = bf2f(P[base + (size_t)(t0 - d) * D]);
    Bsum += x;
    Asum += (float)d * x;
  }
  for (int t = t0; t < t0 + CH; ++t) {
    float xt = bf2f(P[base + (size_t)t * D]);
    P2[base + (size_t)t * D] = f2bf(xt + Asum);
    float x31 = (t >= 30) ? bf2f(P[base + (size_t)(t - 30) * D]) : 0.0f;
    Bsum += xt - x31;
    Asum += Bsum - 30.0f * x31;
  }
}

// ---------------------------------------------------------------------------
extern "C" void kernel_launch(void* const* d_in, const int* in_sizes, int n_in,
                              void* d_out, int out_size, void* d_ws,
                              size_t ws_size, hipStream_t stream) {
  (void)in_sizes; (void)n_in; (void)out_size; (void)ws_size;
  const float* X   = (const float*)d_in[0];  // [8,2048,1024]
  const float* pad = (const float*)d_in[1];  // [8,2048]
  const float* W1  = (const float*)d_in[2];  // [1024,1024]
  const float* b1  = (const float*)d_in[3];  // [1024]
  const float* W2  = (const float*)d_in[4];  // [1024,1024]
  const float* b2  = (const float*)d_in[5];  // [1024]
  float* out = (float*)d_out;                // [8,2048,1024] f32

  unsigned short* P  = (unsigned short*)d_ws;     // 16384x1024 bf16 (32 MB)
  unsigned short* P2 = P + (size_t)16384 * 1024;  // 16384x1024 bf16 (32 MB)

  dim3 gg(1024 / 128, 16384 / 256);  // (N tiles, M tiles) = (8, 64)
  dim3 bb(256);

  fsmn_gemm<0><<<gg, bb, 0, stream>>>(X, nullptr, W1, b1, pad, P, nullptr);
  banded_conv<<<dim3(4, 16, 8), 256, 0, stream>>>(P, P2);
  fsmn_gemm<1><<<gg, bb, 0, stream>>>(nullptr, P2, W2, b2, nullptr, nullptr, out);
}
